// SpatialAttention_83288005804461
// MI455X (gfx1250) — matrix operand order repaired and checked by reading, not compiled, not run on hardware
//
#include <hip/hip_runtime.h>
#include <math.h>

typedef __bf16 bf16_t;
typedef bf16_t v2bf  __attribute__((ext_vector_type(2)));
typedef bf16_t v8bf  __attribute__((ext_vector_type(8)));
typedef bf16_t v16bf __attribute__((ext_vector_type(16)));
typedef float  v8f   __attribute__((ext_vector_type(8)));

#define NHEAD 6
#define POSD  12

// f32 -> bf16 via native hardware convert (RNE)
__device__ __forceinline__ unsigned short f2bf(float f) {
  return __builtin_bit_cast(unsigned short, (bf16_t)f);
}

// pack two f32 into one dword of bf16x2 (lets clang use v_cvt_pk_bf16_f32)
__device__ __forceinline__ unsigned int pk2bf(float x, float y) {
  v2bf t;
  t[0] = (bf16_t)x;
  t[1] = (bf16_t)y;
  return __builtin_bit_cast(unsigned int, t);
}

__device__ __forceinline__ v16bf ld16(const unsigned short* p) {
  return *(const v16bf*)p;
}
__device__ __forceinline__ v16bf ld8x2(const unsigned short* p0,
                                       const unsigned short* p1) {
  v8bf lo = *(const v8bf*)p0;
  v8bf hi = *(const v8bf*)p1;
  return __builtin_shufflevector(lo, hi,
      0,1,2,3,4,5,6,7,8,9,10,11,12,13,14,15);
}

// ---------------------------------------------------------------------------
// Tiny dynamic relative-position-bias MLP: (225,2) -> (225,6) table.
// ---------------------------------------------------------------------------
__device__ __forceinline__ void ln_relu12(float* x, const float* g, const float* b) {
  float m = 0.f;
  #pragma unroll
  for (int c = 0; c < POSD; ++c) m += x[c];
  m *= (1.0f / POSD);
  float v = 0.f;
  #pragma unroll
  for (int c = 0; c < POSD; ++c) { float d = x[c] - m; v += d * d; }
  v *= (1.0f / POSD);
  float inv = rsqrtf(v + 1e-5f);
  #pragma unroll
  for (int c = 0; c < POSD; ++c) {
    float t = (x[c] - m) * inv * g[c] + b[c];
    x[c] = t > 0.f ? t : 0.f;
  }
}

__device__ __forceinline__ void mat12(const float* x, const float* w,
                                      const float* b, float* y) {
  #pragma unroll
  for (int c = 0; c < POSD; ++c) {
    float acc = b[c];
    #pragma unroll
    for (int k = 0; k < POSD; ++k) acc += x[k] * w[k * POSD + c];
    y[c] = acc;
  }
}

__global__ void rpe_mlp_kernel(const float* __restrict__ w_proj,
                               const float* __restrict__ b_proj,
                               const float* __restrict__ ln1_g, const float* __restrict__ ln1_b,
                               const float* __restrict__ w1,    const float* __restrict__ b1,
                               const float* __restrict__ ln2_g, const float* __restrict__ ln2_b,
                               const float* __restrict__ w2,    const float* __restrict__ b2,
                               const float* __restrict__ ln3_g, const float* __restrict__ ln3_b,
                               const float* __restrict__ w3,    const float* __restrict__ b3,
                               float* __restrict__ posTab) {
  int r = blockIdx.x * blockDim.x + threadIdx.x;
  if (r >= 225) return;
  float bh = (float)(r / 15 - 7);
  float bw = (float)(r % 15 - 7);
  float x[POSD], y[POSD];
  #pragma unroll
  for (int c = 0; c < POSD; ++c)
    x[c] = bh * w_proj[c] + bw * w_proj[POSD + c] + b_proj[c];
  ln_relu12(x, ln1_g, ln1_b);
  mat12(x, w1, b1, y);
  ln_relu12(y, ln2_g, ln2_b);
  mat12(y, w2, b2, x);
  ln_relu12(x, ln3_g, ln3_b);
  #pragma unroll
  for (int h = 0; h < NHEAD; ++h) {
    float acc = b3[h];
    #pragma unroll
    for (int k = 0; k < POSD; ++k) acc += x[k] * w3[k * NHEAD + h];
    posTab[r * NHEAD + h] = acc;
  }
}

// ---------------------------------------------------------------------------
// Windowed attention: one block of 128 threads (4 waves) per (window, head).
// bf16 WMMA (16x16x32) for QK^T and PV, f32 accumulate, f32 softmax.
// ---------------------------------------------------------------------------
__global__ __launch_bounds__(128)
void attn_win_kernel(const float* __restrict__ q, const float* __restrict__ k,
                     const float* __restrict__ v, const float* __restrict__ posTab,
                     float* __restrict__ out) {
  __shared__ __attribute__((aligned(32))) unsigned short lQ [64 * 32]; // [tok][d]
  __shared__ __attribute__((aligned(32))) unsigned short lK [64 * 32]; // [tok][d]
  __shared__ __attribute__((aligned(32))) unsigned short lVt[32 * 64]; // [d][tok]
  __shared__ __attribute__((aligned(32))) unsigned short sP [64 * 64]; // probs bf16
  __shared__ float sAttn[64 * 66];                                     // padded scores
  __shared__ float sPos[225 * NHEAD];

  const int tid  = threadIdx.x;
  const int wh   = blockIdx.x;
  const int head = wh % NHEAD;
  const int win  = wh / NHEAD;
  const int bi   = win >> 10;        // /1024 windows per image
  const int wr   = (win >> 5) & 31;
  const int wc   = win & 31;

  // ---- stage bias table ----
  for (int i = tid; i < 225 * NHEAD; i += 128) sPos[i] = posTab[i];

  // ---- stage Q,K,V head slice into LDS (bf16) ----
  const int tok = tid >> 1;          // 0..63
  const int d0  = (tid & 1) * 16;    // 0 or 16
  {
    const int R  = wr * 8 + (tok >> 3);
    const int Cc = wc * 8 + (tok & 7);
    const size_t g = ((size_t)((bi * 256 + R) * 256 + Cc)) * 192 + head * 32 + d0;
    const float4* qv = (const float4*)(q + g);
    const float4* kv = (const float4*)(k + g);
    const float4* vv = (const float4*)(v + g);
    const float SCALE = 0.17677669529663687f;  // 32^-0.5
    #pragma unroll
    for (int i = 0; i < 4; ++i) {
      float4 a = qv[i], b = kv[i], c = vv[i];
      int d = d0 + i * 4;
      // packed dword stores: one ds_store_b64 per matrix per float4
      uint2 qp, kp;
      qp.x = pk2bf(a.x * SCALE, a.y * SCALE);
      qp.y = pk2bf(a.z * SCALE, a.w * SCALE);
      kp.x = pk2bf(b.x, b.y);
      kp.y = pk2bf(b.z, b.w);
      *(uint2*)&lQ[tok * 32 + d] = qp;
      *(uint2*)&lK[tok * 32 + d] = kp;
      // V is stored transposed -> scalar stores
      lVt[(d + 0) * 64 + tok] = f2bf(c.x);
      lVt[(d + 1) * 64 + tok] = f2bf(c.y);
      lVt[(d + 2) * 64 + tok] = f2bf(c.z);
      lVt[(d + 3) * 64 + tok] = f2bf(c.w);
    }
  }
  __syncthreads();

  const int wv   = tid >> 5;   // wave id 0..3 -> row slab 16*wv..+15
  const int lane = tid & 31;
  const int m16  = lane & 15;
  const int kg   = lane >> 4;  // K-group / lane-group select

  // ---- S = (Q*scale) K^T : 4 tiles per wave, one 16x16x32 WMMA each ----
  {
    // A: rows of Q; per ISA 16-bit A layout K = {8kg+j, 16+8kg+j}
    v16bf aQ = ld8x2(&lQ[(16 * wv + m16) * 32 + 8 * kg],
                     &lQ[(16 * wv + m16) * 32 + 16 + 8 * kg]);
    #pragma unroll
    for (int t = 0; t < 4; ++t) {
      // B: column m of S = row (16t+m16) of K, d = {8kg+j, 16+8kg+j}
      v16bf bK = ld8x2(&lK[(16 * t + m16) * 32 + 8 * kg],
                       &lK[(16 * t + m16) * 32 + 16 + 8 * kg]);
      v8f c = {};
      c = __builtin_amdgcn_wmma_f32_16x16x32_bf16(false, aQ, false, bK,
                                                  (short)0, c, false, false);
      #pragma unroll
      for (int r = 0; r < 8; ++r)
        sAttn[(16 * wv + 8 * kg + r) * 66 + 16 * t + m16] = c[r];
    }
  }
  __syncthreads();

  // ---- softmax(S + rpb) per row, probs -> bf16 ----
  if (tid < 64) {
    const int n  = tid;
    const int ni = n >> 3, nj = n & 7;
    float* rowp = &sAttn[n * 66];
    float mx = -3.4e38f;
    for (int m = 0; m < 64; ++m) {
      int di = ni - (m >> 3) + 7;
      int dj = nj - (m & 7) + 7;
      float x = rowp[m] + sPos[(di * 15 + dj) * NHEAD + head];
      rowp[m] = x;
      mx = fmaxf(mx, x);
    }
    float s = 0.f;
    for (int m = 0; m < 64; ++m) {
      float e = __expf(rowp[m] - mx);
      rowp[m] = e;
      s += e;
    }
    float inv = 1.0f / s;
    for (int m = 0; m < 64; m += 2)
      *(unsigned int*)&sP[n * 64 + m] = pk2bf(rowp[m] * inv, rowp[m + 1] * inv);
  }
  __syncthreads();

  // ---- O = P V : 2 col-tiles x 2 K-steps per wave ----
  v8f co0 = {}, co1 = {};
  #pragma unroll
  for (int s = 0; s < 2; ++s) {
    v16bf aP = ld8x2(&sP[(16 * wv + m16) * 64 + 32 * s + 8 * kg],
                     &sP[(16 * wv + m16) * 64 + 32 * s + 16 + 8 * kg]);
    v16bf bV0 = ld8x2(&lVt[(0  + m16) * 64 + 32 * s + 8 * kg],
                      &lVt[(0  + m16) * 64 + 32 * s + 16 + 8 * kg]);
    v16bf bV1 = ld8x2(&lVt[(16 + m16) * 64 + 32 * s + 8 * kg],
                      &lVt[(16 + m16) * 64 + 32 * s + 16 + 8 * kg]);
    co0 = __builtin_amdgcn_wmma_f32_16x16x32_bf16(false, aP, false, bV0,
                                                  (short)0, co0, false, false);
    co1 = __builtin_amdgcn_wmma_f32_16x16x32_bf16(false, aP, false, bV1,
                                                  (short)0, co1, false, false);
  }

  // ---- store O to (B, H, W, C) ----
  #pragma unroll
  for (int r = 0; r < 8; ++r) {
    int tok2 = 16 * wv + 8 * kg + r;
    int Ro = wr * 8 + (tok2 >> 3);
    int Co = wc * 8 + (tok2 & 7);
    size_t g = ((size_t)((bi * 256 + Ro) * 256 + Co)) * 192 + head * 32;
    out[g + 0  + m16] = co0[r];
    out[g + 16 + m16] = co1[r];
  }
}

// ---------------------------------------------------------------------------
extern "C" void kernel_launch(void* const* d_in, const int* in_sizes, int n_in,
                              void* d_out, int out_size, void* d_ws, size_t ws_size,
                              hipStream_t stream) {
  const float* q      = (const float*)d_in[0];
  const float* k      = (const float*)d_in[1];
  const float* v      = (const float*)d_in[2];
  // d_in[3], d_in[4] are H, W scalars (256, 256) -- hardcoded in kernels
  const float* w_proj = (const float*)d_in[5];
  const float* b_proj = (const float*)d_in[6];
  const float* ln1_g  = (const float*)d_in[7];
  const float* ln1_b  = (const float*)d_in[8];
  const float* w1     = (const float*)d_in[9];
  const float* b1     = (const float*)d_in[10];
  const float* ln2_g  = (const float*)d_in[11];
  const float* ln2_b  = (const float*)d_in[12];
  const float* w2     = (const float*)d_in[13];
  const float* b2     = (const float*)d_in[14];
  const float* ln3_g  = (const float*)d_in[15];
  const float* ln3_b  = (const float*)d_in[16];
  const float* w3     = (const float*)d_in[17];
  const float* b3     = (const float*)d_in[18];
  float* posTab = (float*)d_ws;
  float* out    = (float*)d_out;

  const int B = in_sizes[0] / (256 * 256 * 192);

  rpe_mlp_kernel<<<1, 256, 0, stream>>>(w_proj, b_proj, ln1_g, ln1_b, w1, b1,
                                        ln2_g, ln2_b, w2, b2, ln3_g, ln3_b,
                                        w3, b3, posTab);

  const int nBlocks = B * 1024 * NHEAD;  // windows * heads
  attn_win_kernel<<<nBlocks, 128, 0, stream>>>(q, k, v, posTab, out);
}
